// AdaptiveSpatialTokenizer_6322191859791
// MI455X (gfx1250) — matrix operand order repaired and checked by reading, not compiled, not run on hardware
//
#include <hip/hip_runtime.h>
#include <math.h>

typedef __bf16 v16bf __attribute__((ext_vector_type(16)));
typedef float  v8f   __attribute__((ext_vector_type(8)));
typedef float  v4f   __attribute__((ext_vector_type(4)));

#define BATCH 64
#define NTOK  4096
#define CDIM  256
#define HDIM  64
#define KTOP  16
#define TOK_PER_BLOCK 128           // 8 waves * 16 tokens
#define TILES_PER_B   (NTOK / TOK_PER_BLOCK)   // 32

__device__ __forceinline__ float gelu_tanh(float v) {
    // jax.nn.gelu default (approximate=True)
    const float k0 = 0.7978845608028654f;   // sqrt(2/pi)
    const float k1 = 0.044715f;
    return 0.5f * v * (1.0f + tanhf(k0 * (v + k1 * v * v * v)));
}

// ---------------------------------------------------------------------------
// Kernel 1: importance scores via bf16 WMMA.
//   scores[b][n] = W2 . GELU(W1^T x[b][n] + b1) + b2
// Block: 256 threads (8 waves). Each wave: one 16-token tile, N=H=64 via
// 4 accumulators of 16x16, K loop over C=256 in 8 steps of 32.
// ---------------------------------------------------------------------------
__global__ __launch_bounds__(256) void ast_score_kernel(
    const float* __restrict__ x,  const float* __restrict__ W1,
    const float* __restrict__ b1, const float* __restrict__ W2,
    const float* __restrict__ b2, float* __restrict__ scores)
{
    // bf16 B-fragments of W1: [kk(8)][t(4)][lane(32)] x 16 bf16 (32B each) = 32 KB
    __shared__ __align__(32) __bf16 w1f[8 * 4 * 32 * 16];

    const int tid = threadIdx.x;

    // Cooperative build of W1 B-fragments (ISA 16-bit B 32x16 layout:
    // lanes 0-15: K=0..7,16..23 of the k-step; lanes 16-31: K=8..15,24..31; N = lane%16).
    for (int s = tid; s < 8 * 4 * 32; s += 256) {
        const int kk   = s >> 7;          // / 128
        const int rem  = s & 127;
        const int t    = rem >> 5;        // h-block
        const int ln   = rem & 31;
        const int h    = t * 16 + (ln & 15);
        const int cb   = kk * 32 + ((ln < 16) ? 0 : 8);
        __bf16* dst = &w1f[s * 16];
#pragma unroll
        for (int j = 0; j < 16; ++j)
            dst[j] = (__bf16)W1[(cb + (j < 8 ? j : j + 8)) * HDIM + h];   // W1 is [C][H]
    }
    __syncthreads();

    const int bid       = blockIdx.x;
    const int b         = bid / TILES_PER_B;
    const int tileBase  = (bid % TILES_PER_B) * TOK_PER_BLOCK;
    const int wave      = tid >> 5;
    const int lane      = tid & 31;
    const int tokenBase = tileBase + wave * 16;
    const int row       = tokenBase + (lane & 15);
    const int aoff      = (lane < 16) ? 0 : 8;      // ISA 16-bit A 16x32 layout

    const float* __restrict__ xrow = x + ((size_t)b * NTOK + row) * CDIM;
    const v16bf* __restrict__ wp = (const v16bf*)w1f;  // 32B fragments

    v8f acc0 = {}, acc1 = {}, acc2 = {}, acc3 = {};

#pragma unroll
    for (int kk = 0; kk < 8; ++kk) {
        const int c0 = kk * 32;
        // A fragment: lane<16 holds K=c0+0..7 and c0+16..23 of its row;
        //             lane>=16 holds K=c0+8..15 and c0+24..31.
        v4f f0 = *(const v4f*)(xrow + c0 + aoff);
        v4f f1 = *(const v4f*)(xrow + c0 + aoff + 4);
        v4f f2 = *(const v4f*)(xrow + c0 + 16 + aoff);
        v4f f3 = *(const v4f*)(xrow + c0 + 16 + aoff + 4);
        v16bf a;
#pragma unroll
        for (int j = 0; j < 4; ++j) {
            a[j]      = (__bf16)f0[j];
            a[4 + j]  = (__bf16)f1[j];
            a[8 + j]  = (__bf16)f2[j];
            a[12 + j] = (__bf16)f3[j];
        }
        const int base = (kk * 4) * 32;
        v16bf bf0 = wp[base + 0 * 32 + lane];
        v16bf bf1 = wp[base + 1 * 32 + lane];
        v16bf bf2 = wp[base + 2 * 32 + lane];
        v16bf bf3 = wp[base + 3 * 32 + lane];
        acc0 = __builtin_amdgcn_wmma_f32_16x16x32_bf16(false, a, false, bf0, (short)0, acc0, false, false);
        acc1 = __builtin_amdgcn_wmma_f32_16x16x32_bf16(false, a, false, bf1, (short)0, acc1, false, false);
        acc2 = __builtin_amdgcn_wmma_f32_16x16x32_bf16(false, a, false, bf2, (short)0, acc2, false, false);
        acc3 = __builtin_amdgcn_wmma_f32_16x16x32_bf16(false, a, false, bf3, (short)0, acc3, false, false);
    }

    // Epilogue: +b1, GELU, dot with W2 across the 64 hidden units.
    // C/D layout: lane holds column n = lane%16; VGPR m holds row m (lanes 0-15)
    // or row m+8 (lanes 16-31). Reduce partial dot across each 16-lane half.
    const int n15 = lane & 15;
    const float w20 = W2[ 0 + n15], w21 = W2[16 + n15];
    const float w22 = W2[32 + n15], w23 = W2[48 + n15];
    const float bb0 = b1[ 0 + n15], bb1 = b1[16 + n15];
    const float bb2 = b1[32 + n15], bb3 = b1[48 + n15];
    const float bias2 = b2[0];

    float outv[8];
#pragma unroll
    for (int m = 0; m < 8; ++m) {
        float p = gelu_tanh(acc0[m] + bb0) * w20
                + gelu_tanh(acc1[m] + bb1) * w21
                + gelu_tanh(acc2[m] + bb2) * w22
                + gelu_tanh(acc3[m] + bb3) * w23;
        p += __shfl_xor(p, 1, 32);
        p += __shfl_xor(p, 2, 32);
        p += __shfl_xor(p, 4, 32);
        p += __shfl_xor(p, 8, 32);
        outv[m] = p + bias2;
    }
    if (n15 == 0) {
        const int mb = (lane >> 4) * 8;   // lanes 0..15 -> rows 0..7, 16..31 -> 8..15
        float* srow = scores + (size_t)b * NTOK + tokenBase + mb;
#pragma unroll
        for (int m = 0; m < 8; ++m) srow[m] = outv[m];
    }
}

// ---------------------------------------------------------------------------
// Kernel 2: per-batch top-16 (descending, ties -> lowest index, matching
// jax.lax.top_k) + gather selected tokens + emit indices as floats.
// One block (256 threads) per batch row.
// ---------------------------------------------------------------------------
__global__ __launch_bounds__(256) void ast_topk_gather_kernel(
    const float* __restrict__ x, const float* __restrict__ scores,
    float* __restrict__ out_sel, float* __restrict__ out_idx)
{
    __shared__ float s[NTOK];
    __shared__ float redv[256];
    __shared__ int   redi[256];
    __shared__ int   topi[KTOP];

    const int b = blockIdx.x;
    const int tid = threadIdx.x;
    const float* __restrict__ srow = scores + (size_t)b * NTOK;

    for (int i = tid; i < NTOK; i += 256) s[i] = srow[i];
    __syncthreads();

    for (int k = 0; k < KTOP; ++k) {
        // blocked ownership: thread t owns [16t, 16t+16); '>' keeps lowest idx
        float best = -INFINITY;
        int   bi   = tid * 16;
        const int base = tid * 16;
#pragma unroll
        for (int j = 0; j < 16; ++j) {
            const float v = s[base + j];
            if (v > best) { best = v; bi = base + j; }
        }
        redv[tid] = best; redi[tid] = bi;
        __syncthreads();
        for (int step = 128; step > 0; step >>= 1) {
            if (tid < step) {
                const float ov = redv[tid + step];
                const int   oi = redi[tid + step];
                if (ov > redv[tid] || (ov == redv[tid] && oi < redi[tid])) {
                    redv[tid] = ov; redi[tid] = oi;
                }
            }
            __syncthreads();
        }
        if (tid == 0) topi[k] = redi[0];
        __syncthreads();
        const int widx = topi[k];
        if (tid == (widx >> 4)) s[widx] = -INFINITY;   // owner removes winner
        __syncthreads();
    }

    // Gather selected tokens: 16 rows x 256 floats.
    for (int k = 0; k < KTOP; ++k) {
        const int idx = topi[k];
        const float* __restrict__ xr = x + ((size_t)b * NTOK + idx) * CDIM;
        float* __restrict__ orow = out_sel + ((size_t)b * KTOP + k) * CDIM;
        orow[tid] = xr[tid];               // 256 threads == CDIM
    }
    if (tid < KTOP) out_idx[b * KTOP + tid] = (float)topi[tid];
}

extern "C" void kernel_launch(void* const* d_in, const int* in_sizes, int n_in,
                              void* d_out, int out_size, void* d_ws, size_t ws_size,
                              hipStream_t stream) {
    const float* x  = (const float*)d_in[0];   // (B, N, C)
    const float* W1 = (const float*)d_in[1];   // (C, H)
    const float* b1 = (const float*)d_in[2];   // (H,)
    const float* W2 = (const float*)d_in[3];   // (H, 1)
    const float* b2 = (const float*)d_in[4];   // (1,)

    float* scores  = (float*)d_ws;                              // B*N floats (1 MB)
    float* out_sel = (float*)d_out;                             // B*K*C floats
    float* out_idx = out_sel + (size_t)BATCH * KTOP * CDIM;     // B*K (indices as f32)

    ast_score_kernel<<<BATCH * TILES_PER_B, 256, 0, stream>>>(x, W1, b1, W2, b2, scores);
    ast_topk_gather_kernel<<<BATCH, 256, 0, stream>>>(x, scores, out_sel, out_idx);
}
